// LSTMCell_74612171866302
// MI455X (gfx1250) — matrix operand order repaired and checked by reading, not compiled, not run on hardware
//
#include <hip/hip_runtime.h>
#include <hip/hip_bf16.h>
#include <math.h>

typedef __attribute__((ext_vector_type(16))) _Float16 v16h;
typedef __attribute__((ext_vector_type(8)))  _Float16 v8h;
typedef __attribute__((ext_vector_type(2)))  __fp16   v2fp16;
typedef __attribute__((ext_vector_type(8)))  float    v8f;
typedef __attribute__((ext_vector_type(4)))  unsigned u32x4;
typedef __attribute__((ext_vector_type(8)))  unsigned u32x8;

#define BATCH      65536
#define HID        128
#define NROWS_BLK  256          // batch rows per block
#define CHUNK      32           // batch rows per m-iteration (2 M-subtiles of 16)
#define MITERS     (NROWS_BLK / CHUNK)
#define KTOT       256          // I + H
#define NTOT       512          // 4 gates * H
#define XS         264          // XH LDS row stride in halves (256 + pad)
#define WSR        264          // W  LDS row stride in halves (256 + 8 pad = 528B)
#define LDS_XH_HALVES (CHUNK * XS)                 // 32*264 = 8448
#define LDS_W_HALVES  (NTOT * WSR)                 // 512*264 = 135168
#define LDS_BYTES     ((LDS_XH_HALVES + LDS_W_HALVES) * 2)  // 287232 B < 320KB

__device__ __forceinline__ v16h cat16(v8h lo, v8h hi) {
    return __builtin_shufflevector(lo, hi, 0,1,2,3,4,5,6,7,8,9,10,11,12,13,14,15);
}

__device__ __forceinline__ float sigmoidf_fast(float x) {
    return 1.0f / (1.0f + __expf(-x));
}

// tanh via one transcendental: tanh(x) = sign(x) * (1 - e) / (1 + e),
// e = exp(-2|x|).  exp arg <= 0 so no overflow; |x| large -> e->0 -> 1.
__device__ __forceinline__ float tanhf_fast(float x) {
    const float e = __expf(-2.0f * fabsf(x));
    const float t = (1.0f - e) / (1.0f + e);
    return copysignf(t, x);
}

// packed f32x2 -> f16x2 (v_cvt_pk_rtz_f16_f32), bit-cast to u32 for stores
__device__ __forceinline__ unsigned pkh(float a, float b) {
    v2fp16 t = __builtin_amdgcn_cvt_pkrtz(a, b);
    return __builtin_bit_cast(unsigned, t);
}

// ---------------------------------------------------------------------------
// Pack 8 f32 weight matrices into one f16 [512][256] row-major matrix:
//   row n (gate = n/128, r = n%128): cols 0..127 = W_x[gate][r,:],
//                                    cols 128..255 = W_h[gate][r,:]
// ---------------------------------------------------------------------------
__global__ void lstm_pack_weights(const float* __restrict__ Wii, const float* __restrict__ Whi,
                                  const float* __restrict__ Wif, const float* __restrict__ Whf,
                                  const float* __restrict__ Wig, const float* __restrict__ Whg,
                                  const float* __restrict__ Wio, const float* __restrict__ Who,
                                  _Float16* __restrict__ Wp) {
    const int idx  = blockIdx.x * blockDim.x + threadIdx.x;  // 0 .. 131071
    const int n    = idx >> 8;       // 0..511
    const int k    = idx & 255;      // 0..255
    const int gate = n >> 7;
    const int r    = n & 127;

    const float* src;
    int off;
    if (k < 128) {
        src = (gate == 0) ? Wii : (gate == 1) ? Wif : (gate == 2) ? Wig : Wio;
        off = r * 128 + k;
    } else {
        src = (gate == 0) ? Whi : (gate == 1) ? Whf : (gate == 2) ? Whg : Who;
        off = r * 128 + (k - 128);
    }
    Wp[(size_t)n * 256 + k] = (_Float16)src[off];
}

// ---------------------------------------------------------------------------
// Fused LSTM cell: WMMA GEMM (f16 in, f32 acc) + bias + activations + state
// update. 256 threads = 8 waves; wave w owns hidden columns [w*16, w*16+16)
// for all 4 gates, so it can fuse the elementwise update locally.
// Packed weights are DMA'd to LDS by the Tensor Data Mover (one D#,
// TENSORcnt-tracked), with hardware row padding matching the LDS stride.
// ---------------------------------------------------------------------------
__global__ void __launch_bounds__(256, 1)
lstm_cell_wmma(const float* __restrict__ x,  const float* __restrict__ h,
               const float* __restrict__ c,
               const _Float16* __restrict__ Wp,
               const float* __restrict__ b_i, const float* __restrict__ b_f,
               const float* __restrict__ b_g, const float* __restrict__ b_o,
               float* __restrict__ h_out, float* __restrict__ c_out) {
    extern __shared__ _Float16 smem[];
    _Float16* XH = smem;                    // [CHUNK][XS]
    _Float16* Wl = smem + LDS_XH_HALVES;    // [NTOT][WSR]

    const int tid    = threadIdx.x;
    const int wave   = tid >> 5;            // 0..7 : hidden-column tile
    const int lane   = tid & 31;
    const int lanelo = lane & 15;
    const int lhi    = lane >> 4;           // 0 or 1 (half-wave)
    const int blockRow = blockIdx.x * NROWS_BLK;

    // ---- TDM: single descriptor DMAs the 512x256 f16 weight tile into LDS.
    // pad_interval code 6 (128 DWORDs = one 512B row) + pad_amount code 3
    // (4 DWORDs = 16B) reproduce the padded LDS row stride of 528B (= WSR*2).
    // Flat LDS pointers are {aperture, offset32}: truncation gives lds_addr.
    if (wave == 0) {
        const unsigned lds_base = (unsigned)(uintptr_t)Wl;
        const unsigned long long ga = (unsigned long long)(uintptr_t)Wp;

        u32x4 g0;
        g0[0] = 1u;                                             // count=1 (valid)
        g0[1] = lds_base;                                       // lds_addr
        g0[2] = (unsigned)ga;                                   // global_addr[31:0]
        g0[3] = ((unsigned)(ga >> 32) & 0x01FFFFFFu) | (2u << 30); // [56:32]|type=2

        u32x8 g1;
        g1[0] = (1u << 16)            // data_size = 2 bytes
              | (1u << 20)            // pad_enable
              | (6u << 22)            // pad_interval: 128 DWORDs (512B)
              | (3u << 25);           // pad_amount: 4 DWORDs (16B)
        g1[1] = (unsigned)KTOT << 16; // tensor_dim0 = 256 (atomic_barrier=0)
        g1[2] = (unsigned)NTOT << 16; // tensor_dim1[15:0] = 512
        g1[3] = (unsigned)KTOT << 16; // tile_dim0 = 256 (tensor_dim1 hi = 0)
        g1[4] = (unsigned)NTOT;       // tile_dim1 = 512, tile_dim2 = 0
        g1[5] = (unsigned)KTOT;       // tensor_dim0_stride = 256
        g1[6] = 0u;                   // stride hi / tensor_dim1_stride lo
        g1[7] = 0u;                   // tensor_dim1_stride hi
        asm volatile("tensor_load_to_lds %0, %1"
                     :: "s"(g0), "s"(g1)
                     : "memory");
    }

    for (int m = 0; m < MITERS; ++m) {
        const int rbase = blockRow + m * CHUNK;

        // ---- stage XH = [x | h] chunk (32 rows x 256 cols) as f16 into LDS
        {
            const int r  = tid >> 3;                  // 0..31
            const int cg = tid & 7;                   // 8 segments of 32 cols
            const float* src = (cg < 4)
                ? (x + (size_t)(rbase + r) * 128 + cg * 32)
                : (h + (size_t)(rbase + r) * 128 + (cg - 4) * 32);
            _Float16* dst = XH + r * XS + cg * 32;
            const float4* s4 = (const float4*)src;
            if (m + 1 < MITERS) {
                __builtin_prefetch(src + CHUNK * 128, 0, 1);  // global_prefetch_b8
            }
#pragma unroll
            for (int q = 0; q < 4; ++q) {
                const float4 a0 = s4[2 * q];
                const float4 a1 = s4[2 * q + 1];
                uint4 u;
                u.x = pkh(a0.x, a0.y);
                u.y = pkh(a0.z, a0.w);
                u.z = pkh(a1.x, a1.y);
                u.w = pkh(a1.z, a1.w);
                *(uint4*)(dst + q * 8) = u;
            }
        }
        if (m == 0 && wave == 0) {
            // weight DMA must land before anyone reads Wl (barrier follows)
            __builtin_amdgcn_s_wait_tensorcnt(0);
        }
        __syncthreads();

        // ---- WMMA main loop: D[b, n] += XH[b, k] * W[n, k]
        v8f acc[2][4];
#pragma unroll
        for (int m2 = 0; m2 < 2; ++m2)
#pragma unroll
            for (int g = 0; g < 4; ++g)
                acc[m2][g] = (v8f){0.f, 0.f, 0.f, 0.f, 0.f, 0.f, 0.f, 0.f};

#pragma unroll
        for (int ks = 0; ks < 8; ++ks) {
            const int k0 = ks * 32;

            // A (16x32 f16): lanes 0-15 row M=lane hold K = k0+[0..7],[16..23];
            //                lanes 16-31 hold K = k0+[8..15],[24..31]
            v16h A[2];
#pragma unroll
            for (int m2 = 0; m2 < 2; ++m2) {
                const _Float16* ap = XH + (m2 * 16 + lanelo) * XS + k0 + lhi * 8;
                const v8h alo = *(const v8h*)ap;
                const v8h ahi = *(const v8h*)(ap + 16);
                A[m2] = cat16(alo, ahi);
            }

            // B (32x16 f16): B[k][n] = W[n][k]; lane holds K = k0+[0..7],[16..23]
            // (lanes 16-31: +8) of W row (n = g*128 + wave*16 + lanelo)
#pragma unroll
            for (int g = 0; g < 4; ++g) {
                const _Float16* wp =
                    Wl + (g * 128 + wave * 16 + lanelo) * WSR + k0 + lhi * 8;
                const v8h blo = *(const v8h*)wp;
                const v8h bhi = *(const v8h*)(wp + 16);
                const v16h Bv = cat16(blo, bhi);
                acc[0][g] = __builtin_amdgcn_wmma_f32_16x16x32_f16(
                    false, A[0], false, Bv, (short)0, acc[0][g], false, false);
                acc[1][g] = __builtin_amdgcn_wmma_f32_16x16x32_f16(
                    false, A[1], false, Bv, (short)0, acc[1][g], false, false);
            }
        }

        // ---- fused epilogue: bias, activations, state update, direct store.
        // D layout: VGPR v, lanes 0-15 -> M=v, lanes 16-31 -> M=v+8; N=lanelo.
        const int j  = wave * 16 + lanelo;            // hidden column
        const float bi = b_i[j], bf = b_f[j], bg = b_g[j], bo = b_o[j];
#pragma unroll
        for (int m2 = 0; m2 < 2; ++m2) {
#pragma unroll
            for (int v = 0; v < 8; ++v) {
                const int row = rbase + m2 * 16 + lhi * 8 + v;
                const size_t off = (size_t)row * HID + j;
                const float ia = acc[m2][0][v] + bi;
                const float fa = acc[m2][1][v] + bf;
                const float ga = acc[m2][2][v] + bg;
                const float oa = acc[m2][3][v] + bo;
                const float it = sigmoidf_fast(ia);
                const float ft = sigmoidf_fast(fa);
                const float gt = tanhf_fast(ga);
                const float ot = sigmoidf_fast(oa);
                const float cn = ft * c[off] + it * gt;
                const float hn = ot * tanhf_fast(cn);
                c_out[off] = cn;
                h_out[off] = hn;
            }
        }
        __syncthreads();   // protect XH before next iteration's fill
    }
}

extern "C" void kernel_launch(void* const* d_in, const int* in_sizes, int n_in,
                              void* d_out, int out_size, void* d_ws, size_t ws_size,
                              hipStream_t stream) {
    (void)in_sizes; (void)n_in; (void)out_size; (void)ws_size;

    const float* x   = (const float*)d_in[0];
    const float* h_t = (const float*)d_in[1];
    const float* c_t = (const float*)d_in[2];
    const float* Wii = (const float*)d_in[3];
    const float* Whi = (const float*)d_in[4];
    const float* bi  = (const float*)d_in[5];
    const float* Wif = (const float*)d_in[6];
    const float* Whf = (const float*)d_in[7];
    const float* bf  = (const float*)d_in[8];
    const float* Wig = (const float*)d_in[9];
    const float* Whg = (const float*)d_in[10];
    const float* bg  = (const float*)d_in[11];
    const float* Wio = (const float*)d_in[12];
    const float* Who = (const float*)d_in[13];
    const float* bo  = (const float*)d_in[14];

    _Float16* Wp = (_Float16*)d_ws;                 // 512*256*2 = 256KB scratch
    float* h_out = (float*)d_out;
    float* c_out = h_out + (size_t)BATCH * HID;

    lstm_pack_weights<<<512, 256, 0, stream>>>(Wii, Whi, Wif, Whf,
                                               Wig, Whg, Wio, Who, Wp);
    lstm_cell_wmma<<<BATCH / NROWS_BLK, 256, LDS_BYTES, stream>>>(
        x, h_t, c_t, Wp, bi, bf, bg, bo, h_out, c_out);
}
